// DeepSpeedSparseSelfAttention_50422916055372
// MI455X (gfx1250) — matrix operand order repaired and checked by reading, not compiled, not run on hardware
//
#include <hip/hip_runtime.h>
#include <hip/hip_bf16.h>

typedef __attribute__((ext_vector_type(16))) _Float16 v16h;
typedef __attribute__((ext_vector_type(8)))  _Float16 v8h;
typedef __attribute__((ext_vector_type(4)))  _Float16 v4h;
typedef __attribute__((ext_vector_type(8)))  float    v8f;
typedef __attribute__((ext_vector_type(4)))  float    v4f;

union V16U { v16h v; v8h h[2]; };

// LDS row strides in halves, chosen so lane address steps are coprime-ish with the
// 64 banks (conflict-free b128 fragment loads) while keeping 16B alignment.
#define QS 72   // q_lds:  64 rows  (144B/row -> 36-dword lane step, full bank cycle)
#define KS 72   // k_lds:  32 rows
#define VS 40   // vt_lds: 64 rows  (80B/row -> 20-dword lane step, full bank cycle)
#define PS 40   // p_lds:  16 rows/wave

// sparse layout: per 64-query window w, attended block-cols = {4w..4w+3} plus
// stripes {4j+3, j != w}; pad to 36 entries with one masked dummy.
__device__ __forceinline__ void get_bc(int it, int w, int bc[2], int dum[2]) {
    for (int hc = 0; hc < 2; ++hc) {
        int idx = it * 2 + hc;
        int col, d = 0;
        if (idx < 4) {
            col = 4 * w + idx;
        } else {
            int t = idx - 4;
            int j = t + ((t >= w) ? 1 : 0);
            if (j > 31) { col = 0; d = 1; }
            else        { col = 4 * j + 3; }
        }
        bc[hc] = col; dum[hc] = d;
    }
}

__global__ __launch_bounds__(128)
void sparse_attn_kernel(const float* __restrict__ Q, const float* __restrict__ K,
                        const float* __restrict__ V, float* __restrict__ O)
{
    __shared__ _Float16 q_lds[64 * QS];      // Q tile   [row][dim], f16, pre-scaled
    __shared__ _Float16 k_lds[32 * KS];      // K chunk  [key][dim], f16
    __shared__ _Float16 vt_lds[64 * VS];     // V chunk  [dim][key], f16 (transposed)
    __shared__ _Float16 p_lds[4][16 * PS];   // per-wave unnormalized probs

    const int tid  = threadIdx.x;
    const int lane = tid & 31;
    const int wid  = tid >> 5;
    const int lhi  = lane >> 4;
    const int llo  = lane & 15;

    const int wg = blockIdx.x;
    const int w  = wg & 31;
    const int bh = wg >> 5;
    const size_t base = (size_t)bh * 2048 * 64;

    const float* Qb = Q + base + (size_t)w * 64 * 64;
    const float* Kb = K + base;
    const float* Vb = V + base;
    float*       Ob = O + base + (size_t)w * 64 * 64;

    // scale = d^-0.5 * log2(e): scores arrive pre-multiplied so exp() is a bare exp2.
    const float scale = 0.125f * 1.44269504088896340736f;

    // ---- load + scale + f16-convert Q tile (64x64) ----
    for (int i = 0; i < 8; ++i) {
        int idx = tid + i * 128;
        int row = idx >> 4, c4 = idx & 15;
        v4f q4 = ((const v4f*)Qb)[idx];
        v4h hq;
        hq.x = (_Float16)(q4.x * scale);
        hq.y = (_Float16)(q4.y * scale);
        hq.z = (_Float16)(q4.z * scale);
        hq.w = (_Float16)(q4.w * scale);
        *(v4h*)&q_lds[row * QS + c4 * 4] = hq;
    }
    __syncthreads();

    // ---- Q A-fragments (16x32 per k-step) ----
    v16h qa[2];
    {
        const int qrow = wid * 16 + llo;
        for (int ks = 0; ks < 2; ++ks) {
            const _Float16* p = &q_lds[qrow * QS + ks * 32 + lhi * 8];
            V16U u;
            u.h[0] = *(const v8h*)p;
            u.h[1] = *(const v8h*)(p + 16);
            qa[ks] = u.v;
        }
    }

    // all-ones B fragment: rowsum(P) = P x J via WMMA (sum lands in every lane of the row)
    v16h vones;
#pragma unroll
    for (int i = 0; i < 16; ++i) vones[i] = (_Float16)1.0f;

    v8f acc[4] = {};
    v8f ssum  = {};

    // per-thread fixed load coordinates (row0, row0+8) x c4
    const int prow0 = tid >> 4;
    const int pc4   = tid & 15;

    // ---- prefetch chunk 0 into registers ----
    v4f pk[4], pv[4];
    int bc[2], dum[2];
    get_bc(0, w, bc, dum);
#pragma unroll
    for (int hc = 0; hc < 2; ++hc)
#pragma unroll
        for (int i2 = 0; i2 < 2; ++i2) {
            int r = prow0 + i2 * 8;
            const float* kp = Kb + (size_t)bc[hc] * 1024 + r * 64 + pc4 * 4;
            const float* vp = Vb + (size_t)bc[hc] * 1024 + r * 64 + pc4 * 4;
            pk[hc * 2 + i2] = *(const v4f*)kp;
            pv[hc * 2 + i2] = *(const v4f*)vp;
        }

    for (int it = 0; it < 18; ++it) {
        int cdum0 = dum[0], cdum1 = dum[1];

        __syncthreads();   // previous iteration's LDS reads complete
        // ---- stage prefetched K/V into LDS (convert to f16; V transposed) ----
#pragma unroll
        for (int hc = 0; hc < 2; ++hc)
#pragma unroll
            for (int i2 = 0; i2 < 2; ++i2) {
                int lr = hc * 16 + prow0 + i2 * 8;
                v4f k4 = pk[hc * 2 + i2];
                v4f w4 = pv[hc * 2 + i2];
                v4h hk;
                hk.x = (_Float16)k4.x; hk.y = (_Float16)k4.y;
                hk.z = (_Float16)k4.z; hk.w = (_Float16)k4.w;
                *(v4h*)&k_lds[lr * KS + pc4 * 4] = hk;
                vt_lds[(pc4 * 4 + 0) * VS + lr] = (_Float16)w4.x;
                vt_lds[(pc4 * 4 + 1) * VS + lr] = (_Float16)w4.y;
                vt_lds[(pc4 * 4 + 2) * VS + lr] = (_Float16)w4.z;
                vt_lds[(pc4 * 4 + 3) * VS + lr] = (_Float16)w4.w;
            }
        __syncthreads();

        // ---- issue next chunk's global loads (hidden behind this chunk's compute) ----
        if (it + 1 < 18) {
            get_bc(it + 1, w, bc, dum);
#pragma unroll
            for (int hc = 0; hc < 2; ++hc)
#pragma unroll
                for (int i2 = 0; i2 < 2; ++i2) {
                    int r = prow0 + i2 * 8;
                    const float* kp = Kb + (size_t)bc[hc] * 1024 + r * 64 + pc4 * 4;
                    const float* vp = Vb + (size_t)bc[hc] * 1024 + r * 64 + pc4 * 4;
                    pk[hc * 2 + i2] = *(const v4f*)kp;
                    pv[hc * 2 + i2] = *(const v4f*)vp;
                }
        }

        // ---- scores: two 16-key halves, k-reduction over 64 dims ----
        v8f c[2];
#pragma unroll
        for (int hc = 0; hc < 2; ++hc) {
            const _Float16* kb = &k_lds[(hc * 16 + llo) * KS + lhi * 8];
            V16U b0, b1;
            b0.h[0] = *(const v8h*)(kb);
            b0.h[1] = *(const v8h*)(kb + 16);
            b1.h[0] = *(const v8h*)(kb + 32);
            b1.h[1] = *(const v8h*)(kb + 48);
            v8f z = {};
            v8f t0 = __builtin_amdgcn_wmma_f32_16x16x32_f16(false, qa[0], false, b0.v,
                                                            (short)0, z, false, false);
            c[hc]  = __builtin_amdgcn_wmma_f32_16x16x32_f16(false, qa[1], false, b1.v,
                                                            (short)0, t0, false, false);
        }

        // ---- unnormalized probs: e = exp2(score*scale*log2e); dummy half -> 0 ----
        const float m0 = cdum0 ? 0.0f : 1.0f;
        const float m1 = cdum1 ? 0.0f : 1.0f;
#pragma unroll
        for (int r = 0; r < 8; ++r) {
            float e0 = __builtin_amdgcn_exp2f(c[0][r]) * m0;
            float e1 = __builtin_amdgcn_exp2f(c[1][r]) * m1;
            int prow = r + 8 * lhi;
            p_lds[wid][prow * PS + llo]      = (_Float16)e0;
            p_lds[wid][prow * PS + 16 + llo] = (_Float16)e1;
        }

        // ---- P @ V, plus rowsum via ones-WMMA ----
        {
            const _Float16* pp = &p_lds[wid][llo * PS + lhi * 8];
            V16U u;
            u.h[0] = *(const v8h*)pp;
            u.h[1] = *(const v8h*)(pp + 16);
            v16h pa = u.v;
            ssum = __builtin_amdgcn_wmma_f32_16x16x32_f16(false, pa, false, vones,
                                                          (short)0, ssum, false, false);
#pragma unroll
            for (int dc = 0; dc < 4; ++dc) {
                const _Float16* vb = &vt_lds[(dc * 16 + llo) * VS + lhi * 8];
                V16U bu;
                bu.h[0] = *(const v8h*)(vb);
                bu.h[1] = *(const v8h*)(vb + 16);
                acc[dc] = __builtin_amdgcn_wmma_f32_16x16x32_f16(false, pa, false, bu.v,
                                                                 (short)0, acc[dc], false, false);
            }
        }
    }

    // ---- normalize + write out ----
#pragma unroll
    for (int dc = 0; dc < 4; ++dc)
#pragma unroll
        for (int r = 0; r < 8; ++r) {
            int row = wid * 16 + r + 8 * lhi;
            Ob[row * 64 + dc * 16 + llo] = acc[dc][r] / ssum[r];
        }
}

extern "C" void kernel_launch(void* const* d_in, const int* in_sizes, int n_in,
                              void* d_out, int out_size, void* d_ws, size_t ws_size,
                              hipStream_t stream) {
    (void)in_sizes; (void)n_in; (void)out_size; (void)d_ws; (void)ws_size;
    const float* q = (const float*)d_in[0];
    const float* k = (const float*)d_in[1];
    const float* v = (const float*)d_in[2];
    float* o = (float*)d_out;
    sparse_attn_kernel<<<dim3(1024), dim3(128), 0, stream>>>(q, k, v, o);
}
